// Attention_52922587021993
// MI455X (gfx1250) — compile-verified, matrix operand order checked by reading, not run on hardware
//
#include <hip/hip_runtime.h>

#define DIMC   1024
#define NHEAD  16
#define BATCH  4
#define SEQ    2048
#define DH     64
#define SCALE_F 0.125f            // (1024/16)^-0.5
#define MASK_VAL -10000000.0f

typedef __attribute__((ext_vector_type(16))) __bf16        v16bf;
typedef __attribute__((ext_vector_type(8)))  float         v8f;
typedef __attribute__((ext_vector_type(4)))  unsigned int  u32x4;

// ---------- bf16 hi/lo split helpers (separate planes) ----------
__device__ __forceinline__ unsigned short f2bf_u(float f) {
    unsigned int u = __float_as_uint(f);
    u += 0x7FFFu + ((u >> 16) & 1u);          // round-to-nearest-even
    return (unsigned short)(u >> 16);
}
__device__ __forceinline__ float bfu2f(unsigned short h) {
    return __uint_as_float(((unsigned int)h) << 16);
}
__device__ __forceinline__ __bf16 bfbits(unsigned short h) {
    return __builtin_bit_cast(__bf16, h);
}

// WMMA operand element i maps to k = (i<8?0:16) + 8*hi16 + (i&7) (CDNA5 16-bit
// 16x32 A / 32x16 B layout, wave32). With a contiguous bf16 plane this is just
// two 16-byte loads per lane -> two global_load_b128, no unpack VALU.
__device__ __forceinline__ v16bf load_bf16_row(const unsigned short* __restrict__ rowp,
                                               int k0, int hi16) {
    union { v16bf v; u32x4 q[2]; } u;
    u.q[0] = *(const u32x4*)(rowp + k0 + 8 * hi16);
    u.q[1] = *(const u32x4*)(rowp + k0 + 16 + 8 * hi16);
    return u.v;
}

__device__ __forceinline__ v8f wmma_bf(v16bf a, v16bf b, v8f c) {
    return __builtin_amdgcn_wmma_f32_16x16x32_bf16(false, a, false, b, (short)0, c,
                                                   false, false);
}
// split-precision product: (ah+al)*(bh+bl) ~ ah*bh + ah*bl + al*bh
__device__ __forceinline__ v8f wmma_split(v16bf ah, v16bf al, v16bf bh, v16bf bl, v8f c) {
    c = wmma_bf(ah, bh, c);
    c = wmma_bf(ah, bl, c);
    c = wmma_bf(al, bh, c);
    return c;
}
__device__ __forceinline__ v8f v8f_zero() {
    v8f z;
#pragma unroll
    for (int e = 0; e < 8; ++e) z[e] = 0.0f;
    return z;
}

// ---------- kernel 0: fp32 -> bf16 hi plane + lo plane ----------
__global__ void pack_split_planes(const float* __restrict__ src,
                                  unsigned short* __restrict__ hip_,
                                  unsigned short* __restrict__ lop_, int n) {
    int i = blockIdx.x * blockDim.x + threadIdx.x;
    if (i < n) {
        float x = src[i];
        unsigned short h = f2bf_u(x);
        hip_[i] = h;
        lop_[i] = f2bf_u(x - bfu2f(h));
    }
}

// ---------- kernel 1: QKV projection ----------
// out[m][o] = sum_k X[m][k]*W[o][k]; Q,K row-major per head, V transposed per head.
__global__ void __launch_bounds__(256)
gemm_qkv(const unsigned short* __restrict__ Xh, const unsigned short* __restrict__ Xl,
         const unsigned short* __restrict__ Wh, const unsigned short* __restrict__ Wl,
         unsigned short* __restrict__ Qh, unsigned short* __restrict__ Ql,
         unsigned short* __restrict__ Kh, unsigned short* __restrict__ Kl,
         unsigned short* __restrict__ Vth, unsigned short* __restrict__ Vtl) {
    const int lane = threadIdx.x & 31;
    const int wave = threadIdx.x >> 5;
    const int col  = lane & 15;
    const int hi16 = lane >> 4;
    const int m0 = (blockIdx.x * 8 + wave) * 16;   // token tile
    const int o0 = blockIdx.y * 64;                // output-feature tile (one head slice)

    v8f acc[4];
#pragma unroll
    for (int t = 0; t < 4; ++t) acc[t] = v8f_zero();

    const unsigned short* arh = Xh + (size_t)(m0 + col) * DIMC;
    const unsigned short* arl = Xl + (size_t)(m0 + col) * DIMC;
    const unsigned short* brh = Wh + (size_t)(o0 + col) * DIMC;
    const unsigned short* brl = Wl + (size_t)(o0 + col) * DIMC;
    for (int k0 = 0; k0 < DIMC; k0 += 32) {
        v16bf ah = load_bf16_row(arh, k0, hi16);
        v16bf al = load_bf16_row(arl, k0, hi16);
#pragma unroll
        for (int t = 0; t < 4; ++t) {
            v16bf bh = load_bf16_row(brh + (size_t)t * 16 * DIMC, k0, hi16);
            v16bf bl = load_bf16_row(brl + (size_t)t * 16 * DIMC, k0, hi16);
            acc[t] = wmma_split(ah, al, bh, bl, acc[t]);
        }
    }
    const int s  = o0 >> 10;          // 0=Q 1=K 2=V
    const int h  = (o0 & 1023) >> 6;
    const int b  = m0 >> 11;
    const int n0 = (m0 & 2047) + 8 * hi16;
    const size_t head = (size_t)(b * NHEAD + h);
    if (s == 2) {
        // V transposed: Vt[d][n] so attention's V^T A-operand is contiguous
        unsigned short* bh_ = Vth + head * DH * SEQ;
        unsigned short* bl_ = Vtl + head * DH * SEQ;
#pragma unroll
        for (int t = 0; t < 4; ++t) {
            const int d = t * 16 + col;
#pragma unroll
            for (int r = 0; r < 8; ++r) {
                float x = acc[t][r];
                unsigned short hh = f2bf_u(x);
                bh_[(size_t)d * SEQ + n0 + r] = hh;
                bl_[(size_t)d * SEQ + n0 + r] = f2bf_u(x - bfu2f(hh));
            }
        }
    } else {
        unsigned short* bh_ = (s == 0 ? Qh : Kh) + head * SEQ * DH;
        unsigned short* bl_ = (s == 0 ? Ql : Kl) + head * SEQ * DH;
#pragma unroll
        for (int t = 0; t < 4; ++t) {
            const int d = t * 16 + col;
#pragma unroll
            for (int r = 0; r < 8; ++r) {
                float x = acc[t][r];
                unsigned short hh = f2bf_u(x);
                bh_[(size_t)(n0 + r) * DH + d] = hh;
                bl_[(size_t)(n0 + r) * DH + d] = f2bf_u(x - bfu2f(hh));
            }
        }
    }
}

// ---------- kernel 2: flash attention (transposed-score trick) ----------
__global__ void __launch_bounds__(128)
attn_kernel(const unsigned short* __restrict__ Qh, const unsigned short* __restrict__ Ql,
            const unsigned short* __restrict__ Kh, const unsigned short* __restrict__ Kl,
            const unsigned short* __restrict__ Vth, const unsigned short* __restrict__ Vtl,
            const int* __restrict__ pm,
            unsigned short* __restrict__ Xh, unsigned short* __restrict__ Xl) {
    __shared__ int smask[SEQ];
    const int lane = threadIdx.x & 31;
    const int wave = threadIdx.x >> 5;
    const int col  = lane & 15;     // this lane's query (within tile) / V^T k-col group
    const int hi16 = lane >> 4;
    const int bid = blockIdx.x;
    const int b = bid >> 9;                       // 512 blocks per batch
    const int h = (bid >> 5) & 15;
    const int q0 = ((bid & 31) * 4 + wave) * 16;  // 16 queries per wave

    for (int i = threadIdx.x; i < SEQ; i += 128) smask[i] = pm[b * SEQ + i];
    __syncthreads();

    const size_t head = (size_t)(b * NHEAD + h);
    const unsigned short* qrh = Qh + (head * SEQ + q0 + col) * DH;
    const unsigned short* qrl = Ql + (head * SEQ + q0 + col) * DH;
    const unsigned short* kbh = Kh + head * SEQ * DH;
    const unsigned short* kbl = Kl + head * SEQ * DH;
    const unsigned short* vbh = Vth + head * DH * SEQ;
    const unsigned short* vbl = Vtl + head * DH * SEQ;

    // Q tile as B-operand (lane col = query), K-dim = head dim (2 steps of 32)
    v16bf qh[2], ql[2];
#pragma unroll
    for (int ds = 0; ds < 2; ++ds) {
        qh[ds] = load_bf16_row(qrh, ds * 32, hi16);
        ql[ds] = load_bf16_row(qrl, ds * 32, hi16);
    }

    float mrow = -3.0e38f, lrow = 0.0f;   // per-lane softmax state for query `col`
    v8f o[4];                             // O^T accum: lane holds d = t*16+r+8*hi16, q = col
#pragma unroll
    for (int t = 0; t < 4; ++t) o[t] = v8f_zero();

    for (int k0 = 0; k0 < SEQ; k0 += 32) {
        // S^T = K . Q^T : st[js][r] = score(key k0+js*16+r+8*hi16, query col)
        v8f st[2];
#pragma unroll
        for (int js = 0; js < 2; ++js) {
            v8f acc = v8f_zero();
            const size_t krow = (size_t)(k0 + js * 16 + col) * DH;
#pragma unroll
            for (int ds = 0; ds < 2; ++ds) {
                v16bf kh = load_bf16_row(kbh + krow, ds * 32, hi16);
                v16bf kl = load_bf16_row(kbl + krow, ds * 32, hi16);
                acc = wmma_split(kh, kl, qh[ds], ql[ds], acc);
            }
            st[js] = acc;
        }
        // scale + key padding mask + running max
        float mx = mrow;
#pragma unroll
        for (int js = 0; js < 2; ++js)
#pragma unroll
            for (int r = 0; r < 8; ++r) {
                float s = st[js][r] * SCALE_F;
                if (smask[k0 + js * 16 + 8 * hi16 + r] > 0) s = MASK_VAL;
                st[js][r] = s;
                mx = fmaxf(mx, s);
            }
        mx = fmaxf(mx, __shfl_xor(mx, 16));       // combine the two key halves
        float alpha = __expf(mrow - mx);
        mrow = mx;
        // exp -> P; st layout already matches the B-operand element order exactly
        v16bf ph, pl;
        float rs = 0.0f;
#pragma unroll
        for (int js = 0; js < 2; ++js)
#pragma unroll
            for (int r = 0; r < 8; ++r) {
                float p = __expf(st[js][r] - mx);
                rs += p;
                unsigned short hb = f2bf_u(p);
                ph[js * 8 + r] = bfbits(hb);
                pl[js * 8 + r] = bfbits(f2bf_u(p - bfu2f(hb)));
            }
        rs += __shfl_xor(rs, 16);
        lrow = lrow * alpha + rs;
        // O^T += V^T . P^T  (4 d-subtiles); rescale is a per-lane scalar
#pragma unroll
        for (int t = 0; t < 4; ++t) {
            o[t] = o[t] * alpha;
            const size_t vrow = (size_t)(t * 16 + col) * SEQ;
            v16bf vh = load_bf16_row(vbh + vrow, k0, hi16);
            v16bf vl = load_bf16_row(vbl + vrow, k0, hi16);
            o[t] = wmma_split(vh, vl, ph, pl, o[t]);
        }
    }
    const float linv = 1.0f / lrow;
    // write x[b][n=q0+col][h*64+d] into hi/lo planes for the output GEMM
    unsigned short* xrh = Xh + ((size_t)(b * SEQ + q0 + col)) * DIMC + h * DH;
    unsigned short* xrl = Xl + ((size_t)(b * SEQ + q0 + col)) * DIMC + h * DH;
#pragma unroll
    for (int t = 0; t < 4; ++t)
#pragma unroll
        for (int r = 0; r < 8; ++r) {
            float x = o[t][r] * linv;
            unsigned short hh = f2bf_u(x);
            xrh[t * 16 + 8 * hi16 + r] = hh;
            xrl[t * 16 + 8 * hi16 + r] = f2bf_u(x - bfu2f(hh));
        }
}

// ---------- kernel 3: output projection + bias (fp32 out) ----------
__global__ void __launch_bounds__(256)
gemm_fc(const unsigned short* __restrict__ Xh, const unsigned short* __restrict__ Xl,
        const unsigned short* __restrict__ Wh, const unsigned short* __restrict__ Wl,
        const float* __restrict__ bias,
        float* __restrict__ out) {
    const int lane = threadIdx.x & 31;
    const int wave = threadIdx.x >> 5;
    const int col  = lane & 15;
    const int hi16 = lane >> 4;
    const int m0 = (blockIdx.x * 8 + wave) * 16;
    const int o0 = blockIdx.y * 64;

    v8f acc[4];
#pragma unroll
    for (int t = 0; t < 4; ++t) acc[t] = v8f_zero();

    const unsigned short* arh = Xh + (size_t)(m0 + col) * DIMC;
    const unsigned short* arl = Xl + (size_t)(m0 + col) * DIMC;
    const unsigned short* brh = Wh + (size_t)(o0 + col) * DIMC;
    const unsigned short* brl = Wl + (size_t)(o0 + col) * DIMC;
    for (int k0 = 0; k0 < DIMC; k0 += 32) {
        v16bf ah = load_bf16_row(arh, k0, hi16);
        v16bf al = load_bf16_row(arl, k0, hi16);
#pragma unroll
        for (int t = 0; t < 4; ++t) {
            v16bf bh = load_bf16_row(brh + (size_t)t * 16 * DIMC, k0, hi16);
            v16bf bl = load_bf16_row(brl + (size_t)t * 16 * DIMC, k0, hi16);
            acc[t] = wmma_split(ah, al, bh, bl, acc[t]);
        }
    }
    const int n0 = m0 + 8 * hi16;
#pragma unroll
    for (int t = 0; t < 4; ++t) {
        const int oc = o0 + t * 16 + col;
        const float bia = bias[oc];
#pragma unroll
        for (int r = 0; r < 8; ++r)
            out[(size_t)(n0 + r) * DIMC + oc] = acc[t][r] + bia;
    }
}

extern "C" void kernel_launch(void* const* d_in, const int* in_sizes, int n_in,
                              void* d_out, int out_size, void* d_ws, size_t ws_size,
                              hipStream_t stream) {
    const float* inputs = (const float*)d_in[0];
    const float* W_qkv  = (const float*)d_in[1];
    const float* W_fc   = (const float*)d_in[2];
    const float* b_fc   = (const float*)d_in[3];
    const int*   pmask  = (const int*)d_in[4];

    // workspace layout (bf16/ushort elements), all 16B-aligned; 176 MB total
    const size_t nQKV = (size_t)BATCH * NHEAD * SEQ * DH;   // 8388608
    const size_t nX   = (size_t)BATCH * SEQ * DIMC;         // 8388608
    const size_t nWq  = (size_t)3 * DIMC * DIMC;            // 3145728
    const size_t nWf  = (size_t)DIMC * DIMC;                // 1048576
    unsigned short* p = (unsigned short*)d_ws;
    unsigned short* Qh  = p; p += nQKV;
    unsigned short* Ql  = p; p += nQKV;
    unsigned short* Kh  = p; p += nQKV;
    unsigned short* Kl  = p; p += nQKV;
    unsigned short* Vth = p; p += nQKV;
    unsigned short* Vtl = p; p += nQKV;
    unsigned short* Xih = p; p += nX;
    unsigned short* Xil = p; p += nX;
    unsigned short* Wqh = p; p += nWq;
    unsigned short* Wql = p; p += nWq;
    unsigned short* Wfh = p; p += nWf;
    unsigned short* Wfl = p; p += nWf;
    unsigned short* Xah = p; p += nX;
    unsigned short* Xal = p; p += nX;

    pack_split_planes<<<((int)nX  + 255) / 256, 256, 0, stream>>>(inputs, Xih, Xil, (int)nX);
    pack_split_planes<<<((int)nWq + 255) / 256, 256, 0, stream>>>(W_qkv,  Wqh, Wql, (int)nWq);
    pack_split_planes<<<((int)nWf + 255) / 256, 256, 0, stream>>>(W_fc,   Wfh, Wfl, (int)nWf);

    gemm_qkv<<<dim3(64, 48), 256, 0, stream>>>(Xih, Xil, Wqh, Wql,
                                               Qh, Ql, Kh, Kl, Vth, Vtl);
    attn_kernel<<<BATCH * NHEAD * (SEQ / 64), 128, 0, stream>>>(Qh, Ql, Kh, Kl, Vth, Vtl,
                                                                pmask, Xah, Xal);
    gemm_fc<<<dim3(64, 16), 256, 0, stream>>>(Xah, Xal, Wfh, Wfl, b_fc, (float*)d_out);
}
